// MiRNADiseaseAssociationPrediction_69303592288614
// MI455X (gfx1250) — matrix operand order repaired and checked by reading, not compiled, not run on hardware
//
#include <hip/hip_runtime.h>
#include <math.h>

// ---------------- problem constants ----------------
#define N_ATOM   8192
#define E_BOND   65536
#define DEG      16
#define DIM      64
#define NSEQ     1024
#define RAD      2
#define LGNN     3
#define LOUT     2
#define NCLS     2

typedef __attribute__((ext_vector_type(16))) __bf16 v16bf;
typedef __attribute__((ext_vector_type(8)))  float  v8f;

__device__ __forceinline__ float sigmoidf_(float v) { return 1.0f / (1.0f + __expf(-v)); }

// WMMA A-operand (16x32 bf16) for this lane from two contiguous 8-float runs.
// lane l: M=l&15, h=l>>4 ; a[j]=A[M][k0+16h+j], a[8+j]=A[M][k0+16h+8+j]
__device__ __forceinline__ v16bf cvt_a(const float* lo, const float* hi) {
  v16bf a;
#pragma unroll
  for (int j = 0; j < 8; ++j) { a[j] = (__bf16)lo[j]; a[8 + j] = (__bf16)hi[j]; }
  return a;
}

__device__ __forceinline__ v16bf cvt_a_sum(const float* lo0, const float* lo1,
                                           const float* hi0, const float* hi1) {
  v16bf a;
#pragma unroll
  for (int j = 0; j < 8; ++j) {
    a[j]     = (__bf16)(lo0[j] + lo1[j]);
    a[8 + j] = (__bf16)(hi0[j] + hi1[j]);
  }
  return a;
}

__device__ __forceinline__ v8f wmma_bf16(v16bf a, v16bf b, v8f c) {
  return __builtin_amdgcn_wmma_f32_16x16x32_bf16(false, a, false, b, (short)0, c, false, false);
}

// ---------------- Tensor Data Mover staging (guarded) ----------------
#if __has_builtin(__builtin_amdgcn_tensor_load_to_lds)
#define HAVE_TDM 1
typedef __attribute__((ext_vector_type(4))) unsigned v4u;
typedef __attribute__((ext_vector_type(8))) int      v8i;
typedef __attribute__((ext_vector_type(4))) int      v4i;

// Load a 2-D bf16 tile [64 rows(cols of B)][64 k] from hsT (row stride 8192 elems)
// into LDS at byte offset lds_off; TDM packs tile rows contiguously -> BT[c][k].
__device__ __forceinline__ void tdm_load_chunk(const __bf16* gsrc, unsigned lds_off) {
  unsigned long long ga = (unsigned long long)(uintptr_t)gsrc;
  v4u g0;
  g0[0] = 1u;                                              // count=1, user mode
  g0[1] = lds_off;                                         // lds_addr (bytes)
  g0[2] = (unsigned)(ga & 0xFFFFFFFFull);                  // global_addr[31:0]
  g0[3] = (unsigned)((ga >> 32) & 0x1FFFFFFull) | (2u << 30); // addr[56:32] | type=2
  v8i g1;
  g1[0] = (int)(1u << 16);                                 // data_size=1 (2 bytes)
  g1[1] = (int)((8192u & 0xFFFFu) << 16);                  // tensor_dim0[15:0]
  g1[2] = (int)(((8192u >> 16) & 0xFFFFu) | (64u << 16));  // dim0 hi | tensor_dim1=64
  g1[3] = (int)(64u << 16);                                // dim1 hi(0) | tile_dim0=64
  g1[4] = (int)64u;                                        // tile_dim1=64, tile_dim2=0
  g1[5] = (int)8192u;                                      // tensor_dim0_stride lo32
  g1[6] = 0;                                               // stride0 hi | stride1 lo
  g1[7] = 0;                                               // stride1 hi
  v4i z4 = {0, 0, 0, 0};
#if __clang_major__ >= 23
  v8i z8 = {0, 0, 0, 0, 0, 0, 0, 0};
  __builtin_amdgcn_tensor_load_to_lds(g0, g1, z4, z4, z8, 0);
#else
  __builtin_amdgcn_tensor_load_to_lds(g0, g1, z4, z4, 0);
#endif
}
#else
#define HAVE_TDM 0
#endif

// ---------------- tiny kernels ----------------

__global__ void embed_kernel(const int* __restrict__ fp, const float* __restrict__ table,
                             float* __restrict__ x) {
  int gid = blockIdx.x * blockDim.x + threadIdx.x;     // N*DIM threads
  int n = gid >> 6, d = gid & 63;
  x[gid] = table[fp[n] * DIM + d];
}

__global__ void bond_init_kernel(const float* __restrict__ bfeat, const float* __restrict__ W,
                                 const float* __restrict__ b, float* __restrict__ bond) {
  int gid = blockIdx.x * blockDim.x + threadIdx.x;     // E*DIM threads
  int e = gid >> 6, d = gid & 63;
  float s = b[d];
#pragma unroll
  for (int k = 0; k < 10; ++k) s += bfeat[e * 10 + k] * W[k * DIM + d];
  bond[gid] = s;
}

// ------- hs = relu(x @ sub_W + sub_b); emitted TRANSPOSED + bf16: hsT[col][row] -------

__global__ void sub_hs_kernel(const float* __restrict__ x, const float* __restrict__ W,
                              const float* __restrict__ bias, __bf16* __restrict__ hsT) {
  __shared__ __attribute__((aligned(32))) __bf16 WT[64][64];   // [col][k]
  const int tid = threadIdx.x, wave = tid >> 5, lane = tid & 31;
  const int h = lane >> 4, m = lane & 15;
  { // stage W transposed->bf16 : 4096 elems / 256 thr = 16 each
    int r = tid >> 2, c0 = (tid & 3) * 16;
    const float* src = W + r * 64 + c0;
#pragma unroll
    for (int j = 0; j < 16; ++j) WT[c0 + j][r] = (__bf16)src[j];
  }
  __syncthreads();
  const int row0 = blockIdx.x * 128 + wave * 16;
  const float* arow = x + (size_t)(row0 + m) * DIM;
  v8f zero = {0.f,0.f,0.f,0.f,0.f,0.f,0.f,0.f};
  v8f acc[4] = {zero, zero, zero, zero};
#pragma unroll
  for (int s = 0; s < 2; ++s) {
    const int k0 = s * 32;
    v16bf a = cvt_a(arow + k0 + h * 16, arow + k0 + 8 + h * 16);
#pragma unroll
    for (int t = 0; t < 4; ++t) {
      v16bf b = *(const v16bf*)&WT[t * 16 + m][k0 + h * 16];
      acc[t] = wmma_bf16(a, b, acc[t]);
    }
  }
#pragma unroll
  for (int t = 0; t < 4; ++t) {
    const int col = t * 16 + m;
    const float bb = bias[col];
#pragma unroll
    for (int v = 0; v < 8; ++v) {
      float val = fmaxf(acc[t][v] + bb, 0.0f);
      hsT[(size_t)col * N_ATOM + row0 + v + h * 8] = (__bf16)val;   // transposed store
    }
  }
}

// ---------------- x += adjacency @ hs  (HBM-bound; 23.3 TB/s target) ----------------
// 128 blocks x 4 waves; B chunks (bf16, pre-transposed) double-buffered in LDS,
// staged by the Tensor Data Mover (pipelined one chunk ahead) when available.

__global__ void adj_gemm_kernel(const float* __restrict__ A, const __bf16* __restrict__ hsT,
                                float* __restrict__ x) {
  __shared__ __attribute__((aligned(32))) __bf16 BT[2][64][64];   // [buf][col][k]
  const int tid = threadIdx.x;                 // 128 threads = 4 waves
  const int wave = tid >> 5, lane = tid & 31;
  const int h = lane >> 4, m = lane & 15;
  const int row0 = blockIdx.x * 64 + wave * 16;
  const int row  = row0 + m;                   // A row this lane streams

  v8f acc[4];                                  // C initialized with x (residual)
#pragma unroll
  for (int t = 0; t < 4; ++t)
#pragma unroll
    for (int v = 0; v < 8; ++v)
      acc[t][v] = x[(size_t)(row0 + v + h * 8) * DIM + t * 16 + m];

#if HAVE_TDM
  const unsigned bt_base = (unsigned)(uintptr_t)&BT[0][0][0];
  if (wave == 0) {
    tdm_load_chunk(hsT, bt_base);              // chunk 0 -> buf 0
    __builtin_amdgcn_s_wait_tensorcnt(0);
  }
  __syncthreads();
#else
  { // fallback: contiguous bf16 copy, 64B per thread
    const int c = tid >> 1, koff = (tid & 1) * 32;
    const uint4* src = (const uint4*)(hsT + (size_t)c * N_ATOM + koff);
    uint4* dst = (uint4*)&BT[0][c][koff];
#pragma unroll
    for (int j = 0; j < 4; ++j) dst[j] = src[j];
  }
  __syncthreads();
#endif

  for (int c = 0; c < N_ATOM / 64; ++c) {
    const int p = c & 1;
    const int kc = c * 64;
#if HAVE_TDM
    if (c + 1 < N_ATOM / 64 && wave == 0)      // prefetch next chunk into other buffer
      tdm_load_chunk(hsT + (size_t)(kc + 64), bt_base + (unsigned)((p ^ 1) * 64 * 64 * 2));
#else
    if (c + 1 < N_ATOM / 64) {
      const int cc = tid >> 1, koff = (tid & 1) * 32;
      const uint4* src = (const uint4*)(hsT + (size_t)cc * N_ATOM + kc + 64 + koff);
      uint4* dst = (uint4*)&BT[p ^ 1][cc][koff];
#pragma unroll
      for (int j = 0; j < 4; ++j) dst[j] = src[j];
    }
#endif
    const float* arow = A + (size_t)row * N_ATOM + kc;
    __builtin_prefetch(arow + 64, 0, 0);       // next chunk of the A stream
#pragma unroll
    for (int s = 0; s < 2; ++s) {
      const int k0 = s * 32;
      v16bf a = cvt_a(arow + k0 + h * 16, arow + k0 + 8 + h * 16);
#pragma unroll
      for (int t = 0; t < 4; ++t) {
        v16bf b = *(const v16bf*)&BT[p][t * 16 + m][k0 + h * 16];
        acc[t] = wmma_bf16(a, b, acc[t]);
      }
    }
#if HAVE_TDM
    if (wave == 0) __builtin_amdgcn_s_wait_tensorcnt(0);
#endif
    __syncthreads();                           // publish next buffer / retire this one
  }
#pragma unroll
  for (int t = 0; t < 4; ++t)
#pragma unroll
    for (int v = 0; v < 8; ++v)
      x[(size_t)(row0 + v + h * 8) * DIM + t * 16 + m] = acc[t][v];
}

// ------- GNN atom update: nf=leaky(concat(an,bn)@W+b); af_out=sigmoid(af_in+sum_deg nf) -------
// one wave per atom: the 16 A-rows are the atom's 16 neighbors (gathered, L2-resident)

__global__ void gnn_atom_kernel(const float* __restrict__ af_in, const float* __restrict__ bond,
                                const int* __restrict__ adl, const int* __restrict__ bdl,
                                const float* __restrict__ W /*128x64*/,
                                const float* __restrict__ bias,
                                float* __restrict__ af_out) {
  __shared__ __attribute__((aligned(32))) __bf16 WT[64][128];   // [col][k]
  const int tid = threadIdx.x, wave = tid >> 5, lane = tid & 31;
  const int h = lane >> 4, m = lane & 15;
  { // stage 128x64 weights transposed: 32 elems each
    int r = tid >> 1, c0 = (tid & 1) * 32;
    const float* src = W + r * 64 + c0;
#pragma unroll
    for (int j = 0; j < 32; ++j) WT[c0 + j][r] = (__bf16)src[j];
  }
  __syncthreads();
  const int n = blockIdx.x * 8 + wave;
  const int a_idx = adl[n * DEG + m];
  const int b_idx = bdl[n * DEG + m];
  const float* ra = af_in + (size_t)a_idx * DIM;
  const float* rb = bond  + (size_t)b_idx * DIM;
  v8f zero = {0.f,0.f,0.f,0.f,0.f,0.f,0.f,0.f};
  v8f acc[4] = {zero, zero, zero, zero};
#pragma unroll
  for (int s = 0; s < 4; ++s) {                       // K=128 in four 32-steps
    const float* base = (s < 2) ? (ra + s * 32) : (rb + (s - 2) * 32);
    v16bf a = cvt_a(base + h * 16, base + 8 + h * 16);
    const int k0 = s * 32;
#pragma unroll
    for (int t = 0; t < 4; ++t) {
      v16bf b = *(const v16bf*)&WT[t * 16 + m][k0 + h * 16];
      acc[t] = wmma_bf16(a, b, acc[t]);
    }
  }
#pragma unroll
  for (int t = 0; t < 4; ++t) {
    const int col = t * 16 + m;
    const float bb = bias[col];
    float p = 0.0f;
#pragma unroll
    for (int v = 0; v < 8; ++v) {
      float val = acc[t][v] + bb;
      p += (val > 0.0f) ? val : 0.01f * val;
    }
    p += __shfl_xor(p, 16);                           // combine the two row halves
    if (h == 0) {
      float old = af_in[(size_t)n * DIM + col];
      af_out[(size_t)n * DIM + col] = sigmoidf_(old + p);
    }
  }
}

// ------- GNN bond update: pair=af[i0]+af[i1]; bond=sigmoid(bond + pair@side_W + b) -------

__global__ void gnn_bond_kernel(const float* __restrict__ af, float* __restrict__ bond,
                                const int* __restrict__ ibj,
                                const float* __restrict__ W /*64x64*/,
                                const float* __restrict__ bias) {
  __shared__ __attribute__((aligned(32))) __bf16 WT[64][64];
  const int tid = threadIdx.x, wave = tid >> 5, lane = tid & 31;
  const int h = lane >> 4, m = lane & 15;
  {
    int r = tid >> 2, c0 = (tid & 3) * 16;
    const float* src = W + r * 64 + c0;
#pragma unroll
    for (int j = 0; j < 16; ++j) WT[c0 + j][r] = (__bf16)src[j];
  }
  __syncthreads();
  const int e0 = blockIdx.x * 128 + wave * 16;
  const int er = e0 + m;
  const int i0 = ibj[er * 2 + 0];
  const int i1 = ibj[er * 2 + 1];
  const float* r0 = af + (size_t)i0 * DIM;
  const float* r1 = af + (size_t)i1 * DIM;
  v8f zero = {0.f,0.f,0.f,0.f,0.f,0.f,0.f,0.f};
  v8f acc[4] = {zero, zero, zero, zero};
#pragma unroll
  for (int s = 0; s < 2; ++s) {
    const int k0 = s * 32;
    v16bf a = cvt_a_sum(r0 + k0 + h * 16,     r1 + k0 + h * 16,
                        r0 + k0 + 8 + h * 16, r1 + k0 + 8 + h * 16);
#pragma unroll
    for (int t = 0; t < 4; ++t) {
      v16bf b = *(const v16bf*)&WT[t * 16 + m][k0 + h * 16];
      acc[t] = wmma_bf16(a, b, acc[t]);
    }
  }
#pragma unroll
  for (int t = 0; t < 4; ++t) {
    const int col = t * 16 + m;
    const float bb = bias[col];
#pragma unroll
    for (int v = 0; v < 8; ++v) {
      size_t off = (size_t)(e0 + v + h * 8) * DIM + col;
      bond[off] = sigmoidf_(bond[off] + acc[t][v] + bb);   // own-row RMW: in-place safe
    }
  }
}

// ---------------- tail kernels ----------------

__global__ void mean_cat_kernel(const float* __restrict__ x, const float* __restrict__ af,
                                float* __restrict__ cat) {
  __shared__ float red[256];
  const int c = blockIdx.x;                 // 0..63
  float s = 0.0f;
  for (int n = threadIdx.x; n < N_ATOM; n += 256)
    s += x[(size_t)n * DIM + c] + af[(size_t)n * DIM + c];
  red[threadIdx.x] = s;
  __syncthreads();
  for (int off = 128; off > 0; off >>= 1) {
    if (threadIdx.x < off) red[threadIdx.x] += red[threadIdx.x + off];
    __syncthreads();
  }
  if (threadIdx.x == 0) cat[c] = red[0] * (1.0f / (float)N_ATOM);
}

__global__ void mirna_kernel(const float* __restrict__ words, const float* __restrict__ fcW,
                             const float* __restrict__ fcb, float* __restrict__ cat) {
  const int d = threadIdx.x;                // 64 threads
  float s = fcb[d];
  for (int k = 0; k < NSEQ; ++k) s += words[k] * fcW[k * DIM + d];
  cat[DIM + d] = s;
}

__global__ void head_kernel(const float* __restrict__ cat, const float* __restrict__ outW,
                            const float* __restrict__ outb, const float* __restrict__ interW,
                            const float* __restrict__ interb, float* __restrict__ out) {
  __shared__ float v[2 * DIM];
  const int t = threadIdx.x;                // 128 threads
  v[t] = cat[t];
  __syncthreads();
  for (int j = 0; j < LOUT; ++j) {
    float s = outb[j * 2 * DIM + t];
    for (int k = 0; k < 2 * DIM; ++k) s += v[k] * outW[(size_t)j * 4 * DIM * DIM + k * 2 * DIM + t];
    __syncthreads();
    v[t] = fmaxf(s, 0.0f);
    __syncthreads();
  }
  if (t < NCLS) {
    float s = interb[t];
    for (int k = 0; k < 2 * DIM; ++k) s += v[k] * interW[k * NCLS + t];
    out[t] = s;
  }
}

// ---------------- launch ----------------

extern "C" void kernel_launch(void* const* d_in, const int* in_sizes, int n_in,
                              void* d_out, int out_size, void* d_ws, size_t ws_size,
                              hipStream_t stream) {
  (void)in_sizes; (void)n_in; (void)out_size; (void)ws_size;
  const int*   fingerprints = (const int*)  d_in[0];
  const int*   adl          = (const int*)  d_in[1];
  const float* bond_feature = (const float*)d_in[2];
  const int*   bdl          = (const int*)  d_in[3];
  const int*   ibj          = (const int*)  d_in[4];
  const float* adjacency    = (const float*)d_in[5];
  const float* words        = (const float*)d_in[6];
  const float* embed_table  = (const float*)d_in[7];
  const float* bond_W       = (const float*)d_in[8];
  const float* bond_b       = (const float*)d_in[9];
  const float* sub_W        = (const float*)d_in[10];
  const float* sub_b        = (const float*)d_in[11];
  const float* neigh_W      = (const float*)d_in[12];
  const float* neigh_b      = (const float*)d_in[13];
  const float* side_W       = (const float*)d_in[14];
  const float* side_b       = (const float*)d_in[15];
  const float* fc_W         = (const float*)d_in[16];
  const float* fc_b         = (const float*)d_in[17];
  const float* out_W        = (const float*)d_in[18];
  const float* out_b        = (const float*)d_in[19];
  const float* inter_W      = (const float*)d_in[20];
  const float* inter_b      = (const float*)d_in[21];

  const size_t ND = (size_t)N_ATOM * DIM;
  float* ws   = (float*)d_ws;
  float* x    = ws;                               // N*DIM f32
  float* af0  = x    + ND;                        // N*DIM f32
  float* af1  = af0  + ND;                        // N*DIM f32
  float* bond = af1  + ND;                        // E*DIM f32
  float* cat  = bond + (size_t)E_BOND * DIM;      // 2*DIM f32
  __bf16* hsT = (__bf16*)(cat + 128);             // DIM x N_ATOM bf16 (16B-aligned)

  // sub_graph: x = embed; RAD x { hsT = relu(x@W+b)^T(bf16); x += A@hs }
  embed_kernel<<<(N_ATOM * DIM) / 256, 256, 0, stream>>>(fingerprints, embed_table, x);
  for (int i = 0; i < RAD; ++i) {
    sub_hs_kernel<<<N_ATOM / 128, 256, 0, stream>>>(x, sub_W + i * DIM * DIM, sub_b + i * DIM, hsT);
    adj_gemm_kernel<<<N_ATOM / 64, 128, 0, stream>>>(adjacency, hsT, x);
  }

  // gnn
  bond_init_kernel<<<(E_BOND * DIM) / 256, 256, 0, stream>>>(bond_feature, bond_W, bond_b, bond);
  const float* afin = x;
  float* bufs[2] = {af0, af1};
  for (int i = 0; i < LGNN; ++i) {
    float* afout = bufs[i & 1];
    gnn_atom_kernel<<<N_ATOM / 8, 256, 0, stream>>>(afin, bond, adl, bdl,
                                                    neigh_W + (size_t)i * 2 * DIM * DIM,
                                                    neigh_b + i * DIM, afout);
    gnn_bond_kernel<<<E_BOND / 128, 256, 0, stream>>>(afout, bond, ibj,
                                                      side_W + (size_t)i * DIM * DIM,
                                                      side_b + i * DIM);
    afin = afout;
  }

  // head
  mean_cat_kernel<<<DIM, 256, 0, stream>>>(x, afin, cat);
  mirna_kernel<<<1, DIM, 0, stream>>>(words, fc_W, fc_b, cat);
  head_kernel<<<1, 2 * DIM, 0, stream>>>(cat, out_W, out_b, inter_W, inter_b, (float*)d_out);
}
